// SelfAttention_4466765988490
// MI455X (gfx1250) — matrix operand order repaired and checked by reading, not compiled, not run on hardware
//
#include <hip/hip_runtime.h>

// ---------------------------------------------------------------------------
// Self-attention, CDNA5 (gfx1250), wave32 WMMA bf16 pipeline.
// B=2, N=2048, D=1024, H=16, DH=64.
// Workspace layout (48 MiB total):
//   xb  [4096,1024] bf16   8 MiB
//   wwb [3072,1024] bf16   6 MiB
//   owb [1024,1024] bf16   2 MiB
//   qh  [B*H,2048,64] bf16 8 MiB (pre-scaled by 1/8)
//   kh  [B*H,2048,64] bf16 8 MiB
//   vT  [B*H,64,2048] bf16 8 MiB (transposed for contiguous B-operand loads)
//   ao  [4096,1024] bf16   8 MiB (attention output, pre-projection)
// ---------------------------------------------------------------------------

typedef __bf16 bf16;
typedef __attribute__((ext_vector_type(16))) __bf16 v16bf;
typedef __attribute__((ext_vector_type(8)))  __bf16 v8bf;
typedef __attribute__((ext_vector_type(8)))  float  v8f;

#define WMMA_BF16(a, b, c) \
  __builtin_amdgcn_wmma_f32_16x16x32_bf16(false, (a), false, (b), (short)0, (c), false, false)

// A operand: 16x32 (MxK), row-major source with leading dim ld.
// ISA layout: lanes 0-15 -> K 0..7 & 16..23 ; lanes 16-31 -> K 8..15 & 24..31.
__device__ __forceinline__ v16bf load_a16x32(const bf16* __restrict__ p, int ld,
                                             int row0, int k0, int lane) {
  const int half = lane >> 4, r = lane & 15;
  const bf16* base = p + (size_t)(row0 + r) * ld + k0;
  v8bf lo = *(const v8bf*)(base + half * 8);
  v8bf hi = *(const v8bf*)(base + 16 + half * 8);
  v16bf f;
#pragma unroll
  for (int i = 0; i < 8; ++i) { f[i] = lo[i]; f[8 + i] = hi[i]; }
  return f;
}

// B operand: 32x16 (KxN) where source is stored "N rows x K cols" row-major
// (i.e. B^T), leading dim ld. Lanes 0-15 hold K 0..7 & 16..23, lanes 16-31 hold 8..15 & 24..31.
__device__ __forceinline__ v16bf load_b32x16(const bf16* __restrict__ p, int ld,
                                             int col0, int k0, int lane) {
  const int half = lane >> 4, c = lane & 15;
  const bf16* base = p + (size_t)(col0 + c) * ld + k0;
  v8bf lo = *(const v8bf*)(base + half * 8);
  v8bf hi = *(const v8bf*)(base + 16 + half * 8);
  v16bf f;
#pragma unroll
  for (int i = 0; i < 8; ++i) { f[i] = lo[i]; f[8 + i] = hi[i]; }
  return f;
}

__global__ void f32_to_bf16_kernel(const float* __restrict__ in,
                                   bf16* __restrict__ out, int n) {
  int i = blockIdx.x * blockDim.x + threadIdx.x;
  if (i < n) out[i] = (bf16)in[i];
}

// Scatter one 16x16 f32 C-fragment into the per-head Q/K/V buffers.
__device__ __forceinline__ void store_qkv_tile(
    const v8f& acc, int m0, int n0, int lane, const float* __restrict__ bias,
    bf16* __restrict__ qh, bf16* __restrict__ kh, bf16* __restrict__ vT) {
  const int c = lane & 15, half = lane >> 4;
  const int n = n0 + c;
  const float bn = bias[n];
  const int which = n >> 10;          // 0=Q, 1=K, 2=V (16-tile never straddles)
  const int within = n & 1023;
  const int h = within >> 6, dh = within & 63;
#pragma unroll
  for (int r = 0; r < 8; ++r) {
    const int m = m0 + r + 8 * half;  // token over B*N
    const int bb = m >> 11, tok = m & 2047;
    const int bh = bb * 16 + h;
    const float v = acc[r] + bn;
    if (which == 0)      qh[((size_t)bh * 2048 + tok) * 64 + dh] = (bf16)(v * 0.125f);
    else if (which == 1) kh[((size_t)bh * 2048 + tok) * 64 + dh] = (bf16)v;
    else                 vT[((size_t)bh * 64 + dh) * 2048 + tok] = (bf16)v;
  }
}

// QKV projection: qkv[m, n] = sum_k x[m,k] * Ww[n,k] + b[n]; scatter to heads.
// Grid (128, 12), block 256 = 8 waves. Each wave computes a 32x32 output tile
// (4 accumulators; A and B fragments each reused twice -> 16 FLOP/byte).
__global__ __launch_bounds__(256) void qkv_gemm_kernel(
    const bf16* __restrict__ xb,   // [4096,1024]
    const bf16* __restrict__ wwb,  // [3072,1024]
    const float* __restrict__ bias,// [3072]
    bf16* __restrict__ qh, bf16* __restrict__ kh, bf16* __restrict__ vT) {
  const int lane = threadIdx.x & 31;
  const int wave = threadIdx.x >> 5;
  const int m0 = blockIdx.x * 32;
  const int n0 = (blockIdx.y * 8 + wave) * 32;
  v8f acc00 = {}, acc01 = {}, acc10 = {}, acc11 = {};
  for (int k = 0; k < 1024; k += 32) {
    __builtin_prefetch(xb + (size_t)(m0 + (lane & 15)) * 1024 + k + 256, 0, 1);
    __builtin_prefetch(wwb + (size_t)(n0 + (lane & 15)) * 1024 + k + 256, 0, 1);
    v16bf a0 = load_a16x32(xb, 1024, m0,      k, lane);
    v16bf a1 = load_a16x32(xb, 1024, m0 + 16, k, lane);
    v16bf b0 = load_b32x16(wwb, 1024, n0,      k, lane);
    v16bf b1 = load_b32x16(wwb, 1024, n0 + 16, k, lane);
    acc00 = WMMA_BF16(a0, b0, acc00);
    acc01 = WMMA_BF16(a0, b1, acc01);
    acc10 = WMMA_BF16(a1, b0, acc10);
    acc11 = WMMA_BF16(a1, b1, acc11);
  }
  store_qkv_tile(acc00, m0,      n0,      lane, bias, qh, kh, vT);
  store_qkv_tile(acc01, m0,      n0 + 16, lane, bias, qh, kh, vT);
  store_qkv_tile(acc10, m0 + 16, n0,      lane, bias, qh, kh, vT);
  store_qkv_tile(acc11, m0 + 16, n0 + 16, lane, bias, qh, kh, vT);
}

// Flash attention: one wave per (b, h, 16 query rows). Online softmax over
// 32-key steps; S and O tiles stay in WMMA accumulators; P goes through LDS
// to re-layout C-fragment -> A-fragment and convert to bf16.
__global__ __launch_bounds__(32) void attn_kernel(
    const bf16* __restrict__ qh, const bf16* __restrict__ kh,
    const bf16* __restrict__ vT, bf16* __restrict__ ao /*[4096,1024]*/) {
  __shared__ __align__(16) bf16 sP[16][32];
  const int lane = threadIdx.x & 31;
  const int i0 = blockIdx.x * 16;
  const int h = blockIdx.y, bb = blockIdx.z;
  const int bh = bb * 16 + h;
  const bf16* q = qh + (size_t)bh * 2048 * 64;
  const bf16* kk = kh + (size_t)bh * 2048 * 64;
  const bf16* vv = vT + (size_t)bh * 64 * 2048;
  const int half = lane >> 4, cc = lane & 15;

  const v16bf aq0 = load_a16x32(q, 64, i0, 0, lane);
  const v16bf aq1 = load_a16x32(q, 64, i0, 32, lane);

  float mr[8], lr[8];
#pragma unroll
  for (int r = 0; r < 8; ++r) { mr[r] = -3.0e38f; lr[r] = 0.0f; }
  v8f o0 = {}, o1 = {}, o2 = {}, o3 = {};

  for (int j = 0; j < 2048; j += 32) {
    // S = (q * 1/sqrt(dh)) . k^T  for key cols [j, j+32)
    v16bf bk0 = load_b32x16(kk, 64, j,      0,  lane);
    v16bf bk1 = load_b32x16(kk, 64, j,      32, lane);
    v16bf bk2 = load_b32x16(kk, 64, j + 16, 0,  lane);
    v16bf bk3 = load_b32x16(kk, 64, j + 16, 32, lane);
    v8f s0 = {}, s1 = {};
    s0 = WMMA_BF16(aq0, bk0, s0);
    s0 = WMMA_BF16(aq1, bk1, s0);
    s1 = WMMA_BF16(aq0, bk2, s1);
    s1 = WMMA_BF16(aq1, bk3, s1);

    // Online softmax. Row m = r + 8*half lives across the 16 lanes of this
    // half at element r, so butterfly within lane-halves reduces per row.
    float mnew[8];
#pragma unroll
    for (int r = 0; r < 8; ++r) {
      float t = fmaxf(s0[r], s1[r]);
#pragma unroll
      for (int msk = 1; msk < 16; msk <<= 1) t = fmaxf(t, __shfl_xor(t, msk, 32));
      mnew[r] = fmaxf(mr[r], t);
    }
#pragma unroll
    for (int r = 0; r < 8; ++r) {
      const float p0 = __expf(s0[r] - mnew[r]);
      const float p1 = __expf(s1[r] - mnew[r]);
      float rs = p0 + p1;
#pragma unroll
      for (int msk = 1; msk < 16; msk <<= 1) rs += __shfl_xor(rs, msk, 32);
      const float alpha = __expf(mr[r] - mnew[r]);
      lr[r] = lr[r] * alpha + rs;
      mr[r] = mnew[r];
      o0[r] *= alpha; o1[r] *= alpha; o2[r] *= alpha; o3[r] *= alpha;
      sP[r + 8 * half][cc]      = (bf16)p0;
      sP[r + 8 * half][16 + cc] = (bf16)p1;
    }
    asm volatile("s_wait_dscnt 0" ::: "memory");

    // O += P . V  (P: 16x32 A-operand from LDS, V^T rows give contiguous B)
    v16bf ap  = load_a16x32(&sP[0][0], 32, 0, 0, lane);
    v16bf bv0 = load_b32x16(vv, 2048, 0,  j, lane);
    v16bf bv1 = load_b32x16(vv, 2048, 16, j, lane);
    v16bf bv2 = load_b32x16(vv, 2048, 32, j, lane);
    v16bf bv3 = load_b32x16(vv, 2048, 48, j, lane);
    o0 = WMMA_BF16(ap, bv0, o0);
    o1 = WMMA_BF16(ap, bv1, o1);
    o2 = WMMA_BF16(ap, bv2, o2);
    o3 = WMMA_BF16(ap, bv3, o3);
  }

#pragma unroll
  for (int r = 0; r < 8; ++r) {
    const float inv = 1.0f / lr[r];
    const int tok = i0 + r + 8 * half;
    const size_t base = ((size_t)bb * 2048 + tok) * 1024 + h * 64;
    ao[base + 0  + cc] = (bf16)(o0[r] * inv);
    ao[base + 16 + cc] = (bf16)(o1[r] * inv);
    ao[base + 32 + cc] = (bf16)(o2[r] * inv);
    ao[base + 48 + cc] = (bf16)(o3[r] * inv);
  }
}

// Output projection: out[m,n] = sum_k ao[m,k] * Wout[n,k] + b[n], f32 result.
// Grid (128, 4), block 256 = 8 waves, each wave a 32x32 output tile.
__global__ __launch_bounds__(256) void out_gemm_kernel(
    const bf16* __restrict__ ao,   // [4096,1024]
    const bf16* __restrict__ owb,  // [1024,1024]
    const float* __restrict__ ob,  // [1024]
    float* __restrict__ out) {     // [4096,1024]
  const int lane = threadIdx.x & 31;
  const int wave = threadIdx.x >> 5;
  const int m0 = blockIdx.x * 32;
  const int n0 = (blockIdx.y * 8 + wave) * 32;
  v8f acc00 = {}, acc01 = {}, acc10 = {}, acc11 = {};
  for (int k = 0; k < 1024; k += 32) {
    __builtin_prefetch(ao + (size_t)(m0 + (lane & 15)) * 1024 + k + 256, 0, 1);
    __builtin_prefetch(owb + (size_t)(n0 + (lane & 15)) * 1024 + k + 256, 0, 1);
    v16bf a0 = load_a16x32(ao, 1024, m0,      k, lane);
    v16bf a1 = load_a16x32(ao, 1024, m0 + 16, k, lane);
    v16bf b0 = load_b32x16(owb, 1024, n0,      k, lane);
    v16bf b1 = load_b32x16(owb, 1024, n0 + 16, k, lane);
    acc00 = WMMA_BF16(a0, b0, acc00);
    acc01 = WMMA_BF16(a0, b1, acc01);
    acc10 = WMMA_BF16(a1, b0, acc10);
    acc11 = WMMA_BF16(a1, b1, acc11);
  }
  const int c = lane & 15, half = lane >> 4;
  const float bn0 = ob[n0 + c];
  const float bn1 = ob[n0 + 16 + c];
#pragma unroll
  for (int r = 0; r < 8; ++r) {
    const int mA = m0 + r + 8 * half;
    const int mB = m0 + 16 + r + 8 * half;
    out[(size_t)mA * 1024 + n0 + c]      = acc00[r] + bn0;
    out[(size_t)mA * 1024 + n0 + 16 + c] = acc01[r] + bn1;
    out[(size_t)mB * 1024 + n0 + c]      = acc10[r] + bn0;
    out[(size_t)mB * 1024 + n0 + 16 + c] = acc11[r] + bn1;
  }
}

extern "C" void kernel_launch(void* const* d_in, const int* in_sizes, int n_in,
                              void* d_out, int out_size, void* d_ws, size_t ws_size,
                              hipStream_t stream) {
  const float* x  = (const float*)d_in[0];   // [2,2048,1024]
  const float* Ww = (const float*)d_in[1];   // [3072,1024]
  const float* Wb = (const float*)d_in[2];   // [3072]
  const float* ow = (const float*)d_in[3];   // [1024,1024]
  const float* ob = (const float*)d_in[4];   // [1024]

  char* ws = (char*)d_ws;
  bf16* xb  = (bf16*)ws; ws += (size_t)4096 * 1024 * 2;
  bf16* wwb = (bf16*)ws; ws += (size_t)3072 * 1024 * 2;
  bf16* owb = (bf16*)ws; ws += (size_t)1024 * 1024 * 2;
  bf16* qh  = (bf16*)ws; ws += (size_t)32 * 2048 * 64 * 2;
  bf16* kh  = (bf16*)ws; ws += (size_t)32 * 2048 * 64 * 2;
  bf16* vT  = (bf16*)ws; ws += (size_t)32 * 64 * 2048 * 2;
  bf16* ao  = (bf16*)ws; ws += (size_t)4096 * 1024 * 2;

  f32_to_bf16_kernel<<<(4096 * 1024 + 255) / 256, 256, 0, stream>>>(x, xb, 4096 * 1024);
  f32_to_bf16_kernel<<<(3072 * 1024 + 255) / 256, 256, 0, stream>>>(Ww, wwb, 3072 * 1024);
  f32_to_bf16_kernel<<<(1024 * 1024 + 255) / 256, 256, 0, stream>>>(ow, owb, 1024 * 1024);

  qkv_gemm_kernel<<<dim3(128, 12), 256, 0, stream>>>(xb, wwb, Wb, qh, kh, vT);
  attn_kernel<<<dim3(128, 16, 2), 32, 0, stream>>>(qh, kh, vT, ao);
  out_gemm_kernel<<<dim3(128, 4), 256, 0, stream>>>(ao, owb, ob, (float*)d_out);
}
